// SpectralBlock_36189394436285
// MI455X (gfx1250) — matrix operand order repaired and checked by reading, not compiled, not run on hardware
//
#include <hip/hip_runtime.h>
#include <math.h>

// ---------------------------------------------------------------------------
// FNO spectral block for MI455X (gfx1250, wave32, WMMA, async global->LDS).
// Truncated-DFT formulation: every stage is a bf16 WMMA GEMM (f32 accum).
//   P : prep — twiddle tables + bf16 transposed dense kernel into workspace
//   K1: W-axis forward DFT (16 ky modes)        -> S1 [B,H,16,C,{re,im}] bf16
//   K2: H-axis forward DFT (32 kx modes)        -> S2 [B,32,16,C,{re,im}] bf16
//   K3: per-mode complex channel mix (w1/w2)    -> S3 [B,32,16,C,{re,im}] bf16
//   K4: H-axis inverse DFT                      -> S4 [B,H,16,C,{re,im}] bf16
//   K5: W-axis inverse synthesis (single K=32 WMMA) + dense GEMM + bias
//       + residual + tanh-GELU                  -> out [B,H,W,C] f32
// B matrices staged in LDS transposed (N-major): fragment loads are two
// contiguous ds_load_b128 per lane. Straight-copy staging uses
// global_load_async_to_lds_b128 (ASYNCcnt) — CDNA5 async path.
// ---------------------------------------------------------------------------

typedef __attribute__((ext_vector_type(16))) __bf16 v16bf;
typedef __attribute__((ext_vector_type(8)))  __bf16 v8bf;
typedef __attribute__((ext_vector_type(2)))  __bf16 v2bf;
typedef __attribute__((ext_vector_type(8)))  float  v8f;

#define PHASE_STEP 0.049087385212340526f   // 2*pi/128

__device__ __forceinline__ v8f wmma_bf16(v16bf a, v16bf b, v8f c) {
  return __builtin_amdgcn_wmma_f32_16x16x32_bf16(false, a, false, b,
                                                 (short)0, c, false, false);
}

// Generic LDS pointer -> 32-bit LDS byte offset (flat LDS addr = aperture|off).
__device__ __forceinline__ uint32_t lds_addr(const void* p) {
  return (uint32_t)(uintptr_t)p;
}

// Async global->LDS 128-bit copy (per-lane). Tracked by ASYNCcnt.
__device__ __forceinline__ void async_copy_b128(uint32_t lds_off,
                                                const void* base, uint32_t boff) {
  asm volatile("global_load_async_to_lds_b128 %0, %1, %2"
               :: "v"(lds_off), "v"(boff), "s"((uint64_t)(uintptr_t)base)
               : "memory");
}
__device__ __forceinline__ void wait_async() {
  asm volatile("s_wait_asynccnt 0x0" ::: "memory");
}

// A fragment (16x32 bf16, row-major LDS tile): two contiguous 8-elem runs.
__device__ __forceinline__ v16bf frag_a_bf16(const __bf16* tile, int ld, int lane) {
  const int r = lane & 15, half = lane >> 4;
  const __bf16* p = tile + r * ld + (half << 3);
  const v8bf lo = *(const v8bf*)(p);
  const v8bf hi = *(const v8bf*)(p + 16);
  v16bf f;
#pragma unroll
  for (int j = 0; j < 8; ++j) { f[j] = lo[j]; f[j + 8] = hi[j]; }
  return f;
}

// A fragment from f32 LDS (converted on load).
__device__ __forceinline__ v16bf frag_a_f32(const float* tile, int ld, int lane) {
  const int r = lane & 15, half = lane >> 4;
  v16bf f;
#pragma unroll
  for (int v = 0; v < 8; ++v) {
    const int kb = ((v & 4) << 2) + ((v & 3) << 1) + (half << 3);
    f[2 * v]     = (__bf16)tile[r * ld + kb];
    f[2 * v + 1] = (__bf16)tile[r * ld + kb + 1];
  }
  return f;
}

// B fragment from TRANSPOSED (N-major) tileT[n][k]: two contiguous 8-elem runs.
__device__ __forceinline__ v16bf frag_b_T(const __bf16* tileT, int ldk, int lane) {
  const int n = lane & 15, half = lane >> 4;
  const __bf16* p = tileT + n * ldk + (half << 3);
  const v8bf lo = *(const v8bf*)(p);
  const v8bf hi = *(const v8bf*)(p + 16);
  v16bf f;
#pragma unroll
  for (int j = 0; j < 8; ++j) { f[j] = lo[j]; f[j + 8] = hi[j]; }
  return f;
}

// ---------------------------------------------------------------------------
// P: precompute twiddle tables + transposed bf16 dense kernel (once/launch).
// T1: [2][16][128]  K1 planes (cos, -sin)
// T2: [3][32][128]  K2 planes (cos, -sin, +sin)
// T4: [3][128][32]  K4 planes (cos, +sin, -sin)
// T5: [128][32]     K5 synthesis (Hermitian 2x + 1/(H*W) folded in)
// WdT:[64][64]      dense kernel transposed, bf16
// ---------------------------------------------------------------------------
__global__ void __launch_bounds__(256) fno_prep(
    const float* __restrict__ dk, __bf16* __restrict__ T1, __bf16* __restrict__ T2,
    __bf16* __restrict__ T4, __bf16* __restrict__ T5, __bf16* __restrict__ WdT) {
  const int gid = blockIdx.x * 256 + threadIdx.x;   // 2048 threads
  for (int i = gid; i < 4096; i += 2048) {
    const int p = i >> 11, r = i & 2047, ky = r >> 7, w = r & 127;
    const float th = PHASE_STEP * (float)((ky * w) & 127);
    T1[i] = (__bf16)(p ? -sinf(th) : cosf(th));
  }
  for (int i = gid; i < 12288; i += 2048) {
    const int p = i >> 12, r = i & 4095, kxi = r >> 7, h = r & 127;
    const int kxv = (kxi < 16) ? kxi : kxi + 96;
    const float th = PHASE_STEP * (float)((kxv * h) & 127);
    const float v = (p == 0) ? cosf(th) : ((p == 1) ? -sinf(th) : sinf(th));
    T2[i] = (__bf16)v;
  }
  for (int i = gid; i < 12288; i += 2048) {
    const int p = i >> 12, r = i & 4095, h = r >> 5, kxi = r & 31;
    const int kxv = (kxi < 16) ? kxi : kxi + 96;
    const float th = PHASE_STEP * (float)((kxv * h) & 127);
    const float v = (p == 0) ? cosf(th) : ((p == 1) ? sinf(th) : -sinf(th));
    T4[i] = (__bf16)v;
  }
  for (int i = gid; i < 4096; i += 2048) {
    const int w = i >> 5, kk = i & 31, ky = kk >> 1;
    const float th = PHASE_STEP * (float)((ky * w) & 127);
    const float s = (ky == 0 ? 1.0f : 2.0f) * (1.0f / 16384.0f);
    T5[i] = (__bf16)((kk & 1) ? (-sinf(th) * s) : (cosf(th) * s));
  }
  for (int i = gid; i < 4096; i += 2048) {
    const int ci = i >> 6, co = i & 63;
    WdT[co * 64 + ci] = (__bf16)dk[i];
  }
}

// ---------------------------------------------------------------------------
// K1: forward DFT along W. Per block: one (b, h0..h0+7) slab, 2 rows/iter.
// Each wave computes re AND im for its row -> packed v2bf stores.
// ---------------------------------------------------------------------------
__global__ void __launch_bounds__(256) fno_fwd_wdft(
    const float* __restrict__ x, const __bf16* __restrict__ T1,
    __bf16* __restrict__ S1) {
  __shared__ __align__(16) __bf16 tw[4096];          // [cos|{-sin}] 16x128 each
  __shared__ __align__(16) __bf16 xbT[2 * 64 * 128]; // 2 rows, [c][w]
  const int tid = threadIdx.x;
  const int b = blockIdx.x >> 4;
  const int h0 = (blockIdx.x & 15) << 3;

  {  // async straight copy of twiddles (8KB)
    const uint32_t lb = lds_addr(tw);
    for (int i = tid; i < 512; i += 256) async_copy_b128(lb + i * 16, T1, i * 16);
  }

  const int lane = tid & 31, wave = tid >> 5;
  const int row = wave >> 2;                 // 0/1: which h row this wave owns
  const int nt = wave & 3;                   // c tile
  const int n = lane & 15, half = lane >> 4;

  for (int hh = 0; hh < 8; hh += 2) {
    if (hh == 0) wait_async();
    __syncthreads();
    // stage 2 rows of x, transposed to [c][w]
    const float4* src = (const float4*)(x + (((size_t)b * 128 + h0 + hh) << 13));
    if (hh < 6) __builtin_prefetch(src + 4096 + tid * 16, 0, 0);
#pragma unroll
    for (int j = 0; j < 16; ++j) {
      const int i4 = tid + (j << 8);
      const float4 v = src[i4];
      const int rr = i4 >> 11;               // row 0/1 (2048 float4 per row)
      const int r4 = i4 & 2047;
      const int w = r4 >> 4;
      const int c0 = (r4 & 15) << 2;
      __bf16* dst = xbT + rr * 8192;
      dst[(c0 + 0) * 128 + w] = (__bf16)v.x;
      dst[(c0 + 1) * 128 + w] = (__bf16)v.y;
      dst[(c0 + 2) * 128 + w] = (__bf16)v.z;
      dst[(c0 + 3) * 128 + w] = (__bf16)v.w;
    }
    __syncthreads();

    const int h = h0 + hh + row;
    v8f ar = {}, ai = {};
#pragma unroll
    for (int ks = 0; ks < 4; ++ks) {
      const v16bf ac = frag_a_bf16(tw + ks * 32, 128, lane);          // cos
      const v16bf as = frag_a_bf16(tw + 2048 + ks * 32, 128, lane);   // -sin
      const v16bf bm = frag_b_T(xbT + row * 8192 + (nt * 16) * 128 + ks * 32, 128, lane);
      ar = wmma_bf16(ac, bm, ar);
      ai = wmma_bf16(as, bm, ai);
    }
#pragma unroll
    for (int i = 0; i < 8; ++i) {
      const int ky = i + (half << 3);
      const int c = (nt << 4) + n;
      const size_t idx = (((size_t)b * 128 + h) * 16 + ky) * 64 + c;
      v2bf pr; pr[0] = (__bf16)ar[i]; pr[1] = (__bf16)ai[i];
      *(v2bf*)(S1 + idx * 2) = pr;           // packed b32 store
    }
  }
}

// ---------------------------------------------------------------------------
// K2: forward DFT along H. Per block: one (b, ky).
// Complex GEMM [32 kx x 128 h] x [128 h x 64 c]; twiddles async-staged.
// ---------------------------------------------------------------------------
__global__ void __launch_bounds__(256) fno_fwd_hdft(
    const __bf16* __restrict__ S1, const __bf16* __restrict__ T2,
    __bf16* __restrict__ S2) {
  __shared__ __align__(16) __bf16 A3[3 * 4096];       // cos | -sin | +sin
  __shared__ __align__(16) __bf16 BreT[64 * 128], BimT[64 * 128];
  const int tid = threadIdx.x;
  const int b = blockIdx.x >> 4;
  const int ky = blockIdx.x & 15;

  {  // async straight copy of 24KB twiddles
    const uint32_t lb = lds_addr(A3);
    for (int i = tid; i < 1536; i += 256) async_copy_b128(lb + i * 16, T2, i * 16);
  }
  for (int i = tid; i < 8192; i += 256) {             // transpose stage of S1
    const int h = i >> 6, c = i & 63;
    const v2bf pr = *(const v2bf*)(S1 + ((((size_t)b * 128 + h) * 16 + ky) * 64 + c) * 2);
    BreT[c * 128 + h] = pr[0];
    BimT[c * 128 + h] = pr[1];
  }
  wait_async();
  __syncthreads();

  const int lane = tid & 31, wave = tid >> 5;
  const int mt = wave >> 2, nt = wave & 3;
  v8f ar = {}, ai = {};
#pragma unroll
  for (int ks = 0; ks < 4; ++ks) {
    const v16bf fre  = frag_a_bf16(A3 +        mt * 16 * 128 + ks * 32, 128, lane);
    const v16bf fim  = frag_a_bf16(A3 + 4096 + mt * 16 * 128 + ks * 32, 128, lane);
    const v16bf fimN = frag_a_bf16(A3 + 8192 + mt * 16 * 128 + ks * 32, 128, lane);
    const v16bf bre = frag_b_T(BreT + (nt * 16) * 128 + ks * 32, 128, lane);
    const v16bf bim = frag_b_T(BimT + (nt * 16) * 128 + ks * 32, 128, lane);
    ar = wmma_bf16(fre, bre, ar);  ar = wmma_bf16(fimN, bim, ar);
    ai = wmma_bf16(fre, bim, ai);  ai = wmma_bf16(fim,  bre, ai);
  }
  const int n = lane & 15, half = lane >> 4;
#pragma unroll
  for (int i = 0; i < 8; ++i) {
    const int kx = mt * 16 + i + (half << 3);
    const int c = nt * 16 + n;
    const size_t idx = (((size_t)b * 32 + kx) * 16 + ky) * 64 + c;
    v2bf pr; pr[0] = (__bf16)ar[i]; pr[1] = (__bf16)ai[i];
    *(v2bf*)(S2 + idx * 2) = pr;
  }
}

// ---------------------------------------------------------------------------
// K3: per-mode complex channel mixing. Per block: one (kx, ky) mode.
// ---------------------------------------------------------------------------
__global__ void __launch_bounds__(256) fno_modemix(
    const __bf16* __restrict__ S2,
    const float* __restrict__ w1r, const float* __restrict__ w1i,
    const float* __restrict__ w2r, const float* __restrict__ w2i,
    __bf16* __restrict__ S3) {
  __shared__ __align__(16) __bf16 Xre[32 * 64], Xim[32 * 64], XimN[32 * 64];
  __shared__ __align__(16) __bf16 WreT[64 * 64], WimT[64 * 64];
  const int tid = threadIdx.x;
  const int kxi = blockIdx.x >> 4;
  const int ky = blockIdx.x & 15;
  const float* wr = (kxi < 16) ? w1r : w2r;
  const float* wi = (kxi < 16) ? w1i : w2i;
  const int kxe = kxi & 15;

  for (int i = tid; i < 2048; i += 256) {
    const int bb = i >> 6, ci = i & 63;
    const v2bf pr = *(const v2bf*)(S2 + ((((size_t)bb * 32 + kxi) * 16 + ky) * 64 + ci) * 2);
    const float im = (float)pr[1];
    Xre[i] = pr[0]; Xim[i] = (__bf16)im; XimN[i] = (__bf16)(-im);
  }
  const size_t wbase = ((size_t)(kxe * 16 + ky)) << 12;
  for (int i = tid; i < 4096; i += 256) {
    const int ci = i >> 6, co = i & 63;
    WreT[co * 64 + ci] = (__bf16)wr[wbase + i];
    WimT[co * 64 + ci] = (__bf16)wi[wbase + i];
  }
  __syncthreads();

  const int lane = tid & 31, wave = tid >> 5;
  const int mt = wave >> 2, nt = wave & 3;
  v8f ar = {}, ai = {};
#pragma unroll
  for (int ks = 0; ks < 2; ++ks) {
    const v16bf xre  = frag_a_bf16(Xre  + mt * 16 * 64 + ks * 32, 64, lane);
    const v16bf xim  = frag_a_bf16(Xim  + mt * 16 * 64 + ks * 32, 64, lane);
    const v16bf ximN = frag_a_bf16(XimN + mt * 16 * 64 + ks * 32, 64, lane);
    const v16bf wre = frag_b_T(WreT + (nt * 16) * 64 + ks * 32, 64, lane);
    const v16bf wim = frag_b_T(WimT + (nt * 16) * 64 + ks * 32, 64, lane);
    ar = wmma_bf16(xre, wre, ar);  ar = wmma_bf16(ximN, wim, ar);
    ai = wmma_bf16(xim, wre, ai);  ai = wmma_bf16(xre,  wim, ai);
  }
  const int n = lane & 15, half = lane >> 4;
#pragma unroll
  for (int i = 0; i < 8; ++i) {
    const int bb = mt * 16 + i + (half << 3);
    const int co = nt * 16 + n;
    const size_t idx = (((size_t)bb * 32 + kxi) * 16 + ky) * 64 + co;
    v2bf pr; pr[0] = (__bf16)ar[i]; pr[1] = (__bf16)ai[i];
    *(v2bf*)(S3 + idx * 2) = pr;
  }
}

// ---------------------------------------------------------------------------
// K4: inverse DFT along H. Per block: one (b, ky).
// Output S4 [b][h][ky][c][{re,im}] with packed b32 stores.
// ---------------------------------------------------------------------------
__global__ void __launch_bounds__(256) fno_inv_hdft(
    const __bf16* __restrict__ S3, const __bf16* __restrict__ T4,
    __bf16* __restrict__ S4) {
  __shared__ __align__(16) __bf16 A3[3 * 4096];       // cos | +sin | -sin
  __shared__ __align__(16) __bf16 BreT[64 * 32], BimT[64 * 32];
  const int tid = threadIdx.x;
  const int b = blockIdx.x >> 4;
  const int ky = blockIdx.x & 15;

  {  // async straight copy of 24KB twiddles
    const uint32_t lb = lds_addr(A3);
    for (int i = tid; i < 1536; i += 256) async_copy_b128(lb + i * 16, T4, i * 16);
  }
  for (int i = tid; i < 2048; i += 256) {
    const int kx = i >> 6, c = i & 63;
    const v2bf pr = *(const v2bf*)(S3 + ((((size_t)b * 32 + kx) * 16 + ky) * 64 + c) * 2);
    BreT[c * 32 + kx] = pr[0];
    BimT[c * 32 + kx] = pr[1];
  }
  wait_async();
  __syncthreads();

  const int lane = tid & 31, wave = tid >> 5;
  const int mt = wave;                        // h tile
  const v16bf fre  = frag_a_bf16(A3 +        mt * 16 * 32, 32, lane);
  const v16bf fim  = frag_a_bf16(A3 + 4096 + mt * 16 * 32, 32, lane);
  const v16bf fimN = frag_a_bf16(A3 + 8192 + mt * 16 * 32, 32, lane);
  const int n = lane & 15, half = lane >> 4;
#pragma unroll
  for (int nt = 0; nt < 4; ++nt) {
    const v16bf bre = frag_b_T(BreT + (nt * 16) * 32, 32, lane);
    const v16bf bim = frag_b_T(BimT + (nt * 16) * 32, 32, lane);
    v8f ar = {}, ai = {};
    ar = wmma_bf16(fre, bre, ar);  ar = wmma_bf16(fimN, bim, ar);  // c*re - s*im
    ai = wmma_bf16(fim, bre, ai);  ai = wmma_bf16(fre,  bim, ai);  // s*re + c*im
#pragma unroll
    for (int i = 0; i < 8; ++i) {
      const int h = mt * 16 + i + (half << 3);
      const int c = nt * 16 + n;
      const size_t idx = (((size_t)b * 128 + h) * 16 + ky) * 64 + c;
      v2bf pr; pr[0] = (__bf16)ar[i]; pr[1] = (__bf16)ai[i];
      *(v2bf*)(S4 + idx * 2) = pr;
    }
  }
}

// ---------------------------------------------------------------------------
// K5: fused irfft-W synthesis + dense + bias + residual + tanh-GELU.
// x row / synthesis table / dense weights staged via async b128 copies.
// ---------------------------------------------------------------------------
__global__ void __launch_bounds__(256) fno_final(
    const float* __restrict__ x, const __bf16* __restrict__ S4,
    const __bf16* __restrict__ T5, const __bf16* __restrict__ WdTg,
    const float* __restrict__ db, float* __restrict__ out) {
  __shared__ __align__(16) float  xb[128 * 64];
  __shared__ __align__(16) __bf16 A1[128 * 32];
  __shared__ __align__(16) __bf16 BsT[64 * 32];
  __shared__ __align__(16) __bf16 WdT[64 * 64];
  __shared__ float bias[64];
  const int tid = threadIdx.x;
  const int b = blockIdx.x >> 7;
  const int h = blockIdx.x & 127;
  const size_t bh = (size_t)b * 128 + h;

  {  // async straight copies: x row (32KB), A1 (8KB), WdT (8KB)
    const float* xrow = x + (bh << 13);
    const uint32_t lx = lds_addr(xb);
    for (int i = tid; i < 2048; i += 256) async_copy_b128(lx + i * 16, xrow, i * 16);
    const uint32_t la = lds_addr(A1);
    for (int i = tid; i < 512; i += 256) async_copy_b128(la + i * 16, T5, i * 16);
    const uint32_t lw = lds_addr(WdT);
    for (int i = tid; i < 512; i += 256) async_copy_b128(lw + i * 16, WdTg, i * 16);
  }
  // transpose-stage the spectral B tile: S4[bh][ky][c][{re,im}] -> BsT[c][2ky+p]
  for (int j = tid; j < 1024; j += 256) {
    const int ky = j >> 6, c = j & 63;
    const v2bf pr = *(const v2bf*)(S4 + (bh << 11) + (size_t)j * 2);
    *(v2bf*)(&BsT[c * 32 + 2 * ky]) = pr;    // re,im adjacent -> b32 LDS store
  }
  if (tid < 64) bias[tid] = db[tid];
  wait_async();
  __syncthreads();

  const int lane = tid & 31, wave = tid >> 5;
  const int mt = wave;                        // w tile
  const v16bf fA1 = frag_a_bf16(A1 + mt * 16 * 32, 32, lane);
  const v16bf fx0 = frag_a_f32(xb + mt * 16 * 64, 64, lane);
  const v16bf fx1 = frag_a_f32(xb + mt * 16 * 64 + 32, 64, lane);
  const int n = lane & 15, half = lane >> 4;
#pragma unroll
  for (int nt = 0; nt < 4; ++nt) {
    v8f acc = {};
    acc = wmma_bf16(fA1, frag_b_T(BsT + (nt * 16) * 32, 32, lane), acc);       // y_s
    acc = wmma_bf16(fx0, frag_b_T(WdT + (nt * 16) * 64, 64, lane), acc);       // y_p lo
    acc = wmma_bf16(fx1, frag_b_T(WdT + (nt * 16) * 64 + 32, 64, lane), acc);  // y_p hi
#pragma unroll
    for (int i = 0; i < 8; ++i) {
      const int w = mt * 16 + i + (half << 3);
      const int c = nt * 16 + n;
      const float v = acc[i] + xb[w * 64 + c] + bias[c];          // x + y_s + y_p
      const float g = 0.5f * v *
          (1.0f + tanhf(0.7978845608f * (v + 0.044715f * v * v * v)));
      out[((bh << 7) + w) * 64 + c] = g;
    }
  }
}

// ---------------------------------------------------------------------------
// Host launcher
// ---------------------------------------------------------------------------
extern "C" void kernel_launch(void* const* d_in, const int* in_sizes, int n_in,
                              void* d_out, int out_size, void* d_ws, size_t ws_size,
                              hipStream_t stream) {
  const float* x   = (const float*)d_in[0];
  const float* w1r = (const float*)d_in[1];
  const float* w1i = (const float*)d_in[2];
  const float* w2r = (const float*)d_in[3];
  const float* w2i = (const float*)d_in[4];
  const float* dk  = (const float*)d_in[5];
  const float* db  = (const float*)d_in[6];
  float* out = (float*)d_out;

  // Workspace (bf16 elements): ~41.9 MB total.
  __bf16* S1  = (__bf16*)d_ws;       // 8,388,608  [B,H,16,C,{re,im}]
  __bf16* S2  = S1 + 8388608;        // 2,097,152  [B,32,16,C,{re,im}]
  __bf16* S3  = S2 + 2097152;        // 2,097,152
  __bf16* S4  = S3 + 2097152;        // 8,388,608  [B,H,16,C,{re,im}]
  __bf16* T1  = S4 + 8388608;        // 4,096
  __bf16* T2  = T1 + 4096;           // 12,288
  __bf16* T4  = T2 + 12288;          // 12,288
  __bf16* T5  = T4 + 12288;          // 4,096
  __bf16* WdT = T5 + 4096;           // 4,096

  fno_prep    <<<dim3(8),    dim3(256), 0, stream>>>(dk, T1, T2, T4, T5, WdT);
  fno_fwd_wdft<<<dim3(512),  dim3(256), 0, stream>>>(x, T1, S1);
  fno_fwd_hdft<<<dim3(512),  dim3(256), 0, stream>>>(S1, T2, S2);
  fno_modemix <<<dim3(512),  dim3(256), 0, stream>>>(S2, w1r, w1i, w2r, w2i, S3);
  fno_inv_hdft<<<dim3(512),  dim3(256), 0, stream>>>(S3, T4, S4);
  fno_final   <<<dim3(4096), dim3(256), 0, stream>>>(x, S4, T5, WdT, db, out);
}
